// UnrolledRNN_26053271617591
// MI455X (gfx1250) — compile-verified, matrix operand order checked by reading, not run on hardware
//
#include <hip/hip_runtime.h>
#include <math.h>

#define BATCH 256
#define SEQ   2048
#define HID   128
#define BM    16            // rows per block tile
#define RS_U32 68           // h-tile LDS row stride in u32 (136 bf16, 272 B)
#define RS_S16 136
#define XT_S16 (BM * HID)   // 2048 shorts per xh tile (4 KB)

typedef __attribute__((ext_vector_type(16))) __bf16       v16bf;
typedef __attribute__((ext_vector_type(8)))  float        v8f;
typedef __attribute__((ext_vector_type(8)))  unsigned int v8u;

// f32 -> bf16 RNE, result in HIGH 16 bits of the returned u32 (no shift)
__device__ __forceinline__ unsigned bfrnd(float f) {
  unsigned u = __builtin_bit_cast(unsigned, f);
  return u + 0x7fffu + ((u >> 16) & 1u);
}
// pack two rounded values into {hi.bf16, lo.bf16} with ONE v_perm_b32
__device__ __forceinline__ unsigned bfpack(float lo, float hi) {
  return __builtin_amdgcn_perm(bfrnd(hi), bfrnd(lo), 0x07060302u);
}
// f32 -> bf16 bits in low half (for b16 stores)
__device__ __forceinline__ unsigned f2bf(float f) { return bfrnd(f) >> 16; }

// branchless tanh: hw V_TANH if available, else exp2/rcp identity
__device__ __forceinline__ float fast_tanh(float x) {
#if __has_builtin(__builtin_amdgcn_tanhf)
  return __builtin_amdgcn_tanhf(x);
#else
  float e = __builtin_amdgcn_exp2f(x * 2.8853900817779268f);
  return 1.0f - 2.0f * __builtin_amdgcn_rcpf(e + 1.0f);
#endif
}

// 32-bit LDS offset of a shared-memory pointer (addrspace(3) value)
__device__ __forceinline__ unsigned lds_off(const void* p) {
  return (unsigned)(size_t)(__attribute__((address_space(3))) const void*)p;
}

// CDNA5 async DMA: 16 bytes global -> LDS, tracked by ASYNCcnt (ISA 08 §4)
__device__ __forceinline__ void async_b128(unsigned lds_byte_off, const void* gaddr) {
  asm volatile("global_load_async_to_lds_b128 %0, %1, off"
               :: "v"(lds_byte_off), "v"((unsigned long long)(size_t)gaddr)
               : "memory");
}

// Opaque register pin: stops clang from splitting/rematerializing fragments
__device__ __forceinline__ v16bf pin_frag(v8u v) {
  asm volatile("" : "+v"(v));
  return __builtin_bit_cast(v16bf, v);
}

// A-matrix fragment (16x32 bf16) from LDS tile, row-major [16][RS_S16].
// ISA 7.12.2: lane(M)=lane&15; VGPR j holds K pair at
// kOfs(j) = (j<4 ? 2j : 16+2(j-4)) + 8*(lane>>4); as u32: {0,1,2,3,8,9,10,11}+4*hi.
__device__ __forceinline__ v16bf ldsA_frag(const unsigned int* tile, int row, int kk, int hi) {
  const unsigned int* p = tile + row * RS_U32 + kk * 16 + 4 * hi;
  v8u v;
  v[0] = p[0];  v[1] = p[1];  v[2] = p[2];  v[3] = p[3];
  v[4] = p[8];  v[5] = p[9];  v[6] = p[10]; v[7] = p[11];
  return __builtin_bit_cast(v16bf, v);
}

// B-matrix fragment (32x16 bf16) gathered once from global f32 [K=128][N=128].
__device__ __forceinline__ v8u wB_raw(const float* __restrict__ W, int kk, int ncol, int hi) {
  v8u v;
#pragma unroll
  for (int j = 0; j < 8; ++j) {
    int k = kk * 32 + ((j < 4) ? (2 * j) : (16 + 2 * (j - 4))) + 8 * hi;
    v[j] = bfpack(W[(size_t)k * HID + ncol], W[(size_t)(k + 1) * HID + ncol]);
  }
  return v;
}
__device__ __forceinline__ v16bf wB_frag(const float* __restrict__ W, int kk, int ncol, int hi) {
  return __builtin_bit_cast(v16bf, wB_raw(W, kk, ncol, hi));
}

// stage 8 consecutive f32 -> one uint4 of packed bf16
__device__ __forceinline__ uint4 pack8(const float* p) {
  float4 a = *(const float4*)p;
  float4 b = *(const float4*)(p + 4);
  uint4 s;
  s.x = bfpack(a.x, a.y);
  s.y = bfpack(a.z, a.w);
  s.z = bfpack(b.x, b.y);
  s.w = bfpack(b.z, b.w);
  return s;
}

// ---------------- kernel 1: xh = x @ W_xh + b_xh  (bf16 out, full GPU) -------
__global__ __launch_bounds__(256) void xh_gemm(
    const float* __restrict__ x, const float* __restrict__ Wxh,
    const float* __restrict__ bxh, unsigned short* __restrict__ xh) {
  __shared__ __align__(16) unsigned int sx[BM * RS_U32];

  const int tid = threadIdx.x, wave = tid >> 5, lane = tid & 31;
  const int laneM = lane & 15, hi = lane >> 4;
  const int ncol = wave * 16 + laneM;
  const size_t m0 = (size_t)blockIdx.x * BM;        // flattened row of [B*S, 128]
  const int row = tid >> 4, c8 = (tid & 15) * 8;

  // stage 16 x-rows -> bf16 LDS
  *(uint4*)(sx + row * RS_U32 + (tid & 15) * 4) = pack8(x + (m0 + row) * HID + c8);

  v16bf w[4];
#pragma unroll
  for (int kk = 0; kk < 4; ++kk) w[kk] = wB_frag(Wxh, kk, ncol, hi);
  const float bias = bxh[ncol];
  __syncthreads();

  v8f acc;
#pragma unroll
  for (int r = 0; r < 8; ++r) acc[r] = bias;
#pragma unroll
  for (int kk = 0; kk < 4; ++kk)
    acc = __builtin_amdgcn_wmma_f32_16x16x32_bf16(false, ldsA_frag(sx, laneM, kk, hi),
                                                  false, w[kk], (short)0, acc, false, false);

  // transpose through LDS so the 134 MB of bf16 goes out as coalesced b128 stores
  __syncthreads();                                   // all A-frag reads of sx done
  unsigned short* st = (unsigned short*)sx;          // reuse as [16][128] bf16 tile
#pragma unroll
  for (int r = 0; r < 8; ++r)
    st[(r + 8 * hi) * HID + ncol] = (unsigned short)f2bf(acc[r]);
  __syncthreads();
  {
    uint4 v = *(const uint4*)((const unsigned int*)sx + row * (HID / 2) + (tid & 15) * 4);
    *(uint4*)(xh + (m0 + row) * HID + c8) = v;       // 256 B per 16 lanes
  }
}

// ---------------- kernel 2: sequential scan, h = tanh(xh_t + h@W_hh) --------
__global__ __launch_bounds__(256) void rnn_scan(
    const unsigned short* __restrict__ xh, const float* __restrict__ h0,
    const float* __restrict__ Whh, const float* __restrict__ Wout,
    const float* __restrict__ bout, float* __restrict__ out) {
  __shared__ __align__(16) unsigned int smem[2 * BM * RS_U32 + 3 * (XT_S16 / 2)];
  unsigned int*   hB = smem;                                   // 2 h tiles (double buffer)
  unsigned short* xB = (unsigned short*)(smem + 2 * BM * RS_U32); // 3 xh tiles [16][128]

  const int tid = threadIdx.x, wave = tid >> 5, lane = tid & 31;
  const int laneM = lane & 15, hi = lane >> 4;
  const int ncol = wave * 16 + laneM;
  const int b0 = blockIdx.x * BM;

  // weight fragments pinned in VGPRs for the whole scan
  v16bf whh[4];
#pragma unroll
  for (int kk = 0; kk < 4; ++kk) whh[kk] = pin_frag(wB_raw(Whh, kk, ncol, hi));

  // stage h0 -> h buffer 0 (bf16)
  {
    const int row = tid >> 4, scol = (tid & 15) * 8;
    *(uint4*)(hB + row * RS_U32 + (tid & 15) * 4) =
        pack8(h0 + (size_t)(b0 + row) * HID + scol);
  }

  // per-thread async-copy geometry: 16 B of one xh tile per thread per step
  const int crow = tid >> 4;
  const unsigned short* gsrc = xh + (size_t)(b0 + crow) * SEQ * HID + (tid & 15) * 8;
  const unsigned xoff = lds_off(xB) + crow * (HID * 2) + (tid & 15) * 16;

  // prime the pipeline: tiles 0 and 1 in flight
  async_b128(xoff + 0 * (XT_S16 * 2), gsrc);
  async_b128(xoff + 1 * (XT_S16 * 2), gsrc + (size_t)HID);

  for (int t = 0; t < SEQ; ++t) {
    if (t + 1 < SEQ) asm volatile("s_wait_asynccnt 0x1" ::: "memory");
    else             asm volatile("s_wait_asynccnt 0x0" ::: "memory");
    __syncthreads();   // single barrier: tile t landed, h(t) visible, WAR buffers freed
    if (t + 2 < SEQ)   // DMA tile t+2 into the buffer last read at step t-1
      async_b128(xoff + ((t + 2) % 3) * (XT_S16 * 2), gsrc + (size_t)(t + 2) * HID);

    const unsigned int* hR = hB + (t & 1) * (BM * RS_U32);        // h(t)
    unsigned short*     hW = (unsigned short*)(hB + ((t + 1) & 1) * (BM * RS_U32)); // h(t+1)

    // acc seeded from bf16 xh tile (bias folded in by kernel 1)
    const unsigned short* xt = xB + (t % 3) * XT_S16;
    v8f acc;
#pragma unroll
    for (int r = 0; r < 8; ++r)
      acc[r] = __builtin_bit_cast(float, (unsigned)xt[(r + 8 * hi) * HID + ncol] << 16);
#pragma unroll
    for (int kk = 0; kk < 4; ++kk)
      acc = __builtin_amdgcn_wmma_f32_16x16x32_bf16(false, ldsA_frag(hR, laneM, kk, hi),
                                                    false, whh[kk], (short)0, acc, false, false);
#pragma unroll
    for (int r = 0; r < 8; ++r)
      hW[(r + 8 * hi) * RS_S16 + ncol] = (unsigned short)f2bf(fast_tanh(acc[r]));
  }

  // final: out = h_last @ W_out + b_out ; h(SEQ) sits in buffer SEQ&1 == 0
  __syncthreads();
  const unsigned int* hF = hB + (SEQ & 1) * (BM * RS_U32);
  const float bias_o = bout[ncol];
  v8f acc;
#pragma unroll
  for (int r = 0; r < 8; ++r) acc[r] = bias_o;
#pragma unroll
  for (int kk = 0; kk < 4; ++kk)
    acc = __builtin_amdgcn_wmma_f32_16x16x32_bf16(false, ldsA_frag(hF, laneM, kk, hi),
                                                  false, wB_frag(Wout, kk, ncol, hi),
                                                  (short)0, acc, false, false);
#pragma unroll
  for (int r = 0; r < 8; ++r)
    out[(size_t)(b0 + r + 8 * hi) * HID + ncol] = acc[r];
}

extern "C" void kernel_launch(void* const* d_in, const int* in_sizes, int n_in,
                              void* d_out, int out_size, void* d_ws, size_t ws_size,
                              hipStream_t stream) {
  const float* x    = (const float*)d_in[0];
  const float* h0   = (const float*)d_in[1];
  const float* Wxh  = (const float*)d_in[2];
  const float* bxh  = (const float*)d_in[3];
  const float* Whh  = (const float*)d_in[4];
  const float* Wout = (const float*)d_in[5];
  const float* bout = (const float*)d_in[6];
  float* out = (float*)d_out;

  // scratch: xh in bf16, [B][S][H] = 256*2048*128*2 bytes = 128 MiB
  unsigned short* xh = (unsigned short*)d_ws;

  dim3 block(256);
  dim3 grid1((BATCH * SEQ) / BM);   // 32768 blocks: full-GPU GEMM
  hipLaunchKernelGGL(xh_gemm, grid1, block, 0, stream, x, Wxh, bxh, xh);

  dim3 grid2(BATCH / BM);           // 16 blocks carry the recurrence
  hipLaunchKernelGGL(rnn_scan, grid2, block, 0, stream,
                     xh, h0, Whh, Wout, bout, out);
}
